// SelfAttention_67714454388781
// MI455X (gfx1250) — matrix operand order repaired and checked by reading, not compiled, not run on hardware
//
#include <hip/hip_runtime.h>

// ---- portable vector types ----
typedef __attribute__((ext_vector_type(16))) _Float16 v16h;
typedef __attribute__((ext_vector_type(8)))  float    v8f;
typedef __attribute__((ext_vector_type(4)))  unsigned int u32x4;
typedef __attribute__((ext_vector_type(4)))  int      i32x4;
typedef __attribute__((ext_vector_type(4)))  float    f32x4;
typedef __attribute__((ext_vector_type(4)))  _Float16 v4h;

union FragCvt { u32x4 u[2]; v16h h; };
union Half8   { u32x4 u; _Float16 h[8]; };

#define BM 128
#define BN 128
#define BK 32

// CDNA5 async global->LDS (ASYNCcnt-tracked), probe-verified signature:
// (int4 __device__* src, int4 __shared__* dst, imm offset, imm cpol)
#if defined(__gfx1250__) && __has_builtin(__builtin_amdgcn_global_load_async_to_lds_b128)
#define ASYNC_LDS 1
#define NBUF 4   // depth-4 pipeline: 3 tiles in flight behind compute
typedef __attribute__((address_space(1))) i32x4 GlobalI4;
typedef __attribute__((address_space(3))) i32x4 SharedI4;
#else
#define ASYNC_LDS 0
#define NBUF 2
#endif

// C[m,n] = alpha * sum_k A[m,k] * B[n,k]   (both A and B K-contiguous)
//   CT=false: C stored [m][n] (ldc = row stride over n)
//   CT=true : C stored [n][m] (ldc = row stride over m) -> used to emit V^T
template <typename OutT, bool CT>
__global__ __launch_bounds__(256)
void gemm_wmma_f16(const _Float16* __restrict__ A,
                   const _Float16* __restrict__ B,
                   OutT* __restrict__ C,
                   int K, int lda, int ldb, int ldc,
                   long long sA, long long sB, long long sC,
                   float alpha)
{
    __shared__ _Float16 lA[NBUF][BM * BK];
    __shared__ _Float16 lB[NBUF][BN * BK];

    A += (long long)blockIdx.z * sA;
    B += (long long)blockIdx.z * sB;
    C += (long long)blockIdx.z * sC;

    const int tid   = threadIdx.x;
    const int lane  = tid & 31;
    const int wave  = tid >> 5;
    const int waveM = wave & 1;    // 2 slabs of 64 rows
    const int waveN = wave >> 1;   // 4 slabs of 32 cols
    const int m0    = blockIdx.x * BM;
    const int n0    = blockIdx.y * BN;

    v8f acc[4][2];
#pragma unroll
    for (int i = 0; i < 4; ++i)
#pragma unroll
        for (int j = 0; j < 2; ++j)
#pragma unroll
            for (int r = 0; r < 8; ++r) acc[i][j][r] = 0.f;

    // stage1: begin fetching tile (k0) into LDS buffer `buf`
    // (4 async ops per thread per tile: 2 chunks x {A,B})
#if !ASYNC_LDS
    u32x4 ra[2], rb[2];
#endif
    auto stage1 = [&](int buf, int k0) {
#pragma unroll
        for (int i = 0; i < 2; ++i) {
            int chunk = tid + i * 256;         // 512 x 16B chunks per tile
            int r = chunk >> 2, c = chunk & 3; // 4 chunks per 32-half row
            const _Float16* ga = A + (long long)(m0 + r) * lda + (k0 + c * 8);
            const _Float16* gb = B + (long long)(n0 + r) * ldb + (k0 + c * 8);
#if ASYNC_LDS
            __builtin_amdgcn_global_load_async_to_lds_b128(
                (GlobalI4*)(unsigned long long)ga,
                (SharedI4*)&lA[buf][r * BK + c * 8], 0, 0);
            __builtin_amdgcn_global_load_async_to_lds_b128(
                (GlobalI4*)(unsigned long long)gb,
                (SharedI4*)&lB[buf][r * BK + c * 8], 0, 0);
#else
            ra[i] = *(const u32x4*)ga;
            rb[i] = *(const u32x4*)gb;
#endif
        }
    };

    auto compute = [&](int buf) {
        const int rsel = lane & 15;
        const int aoff = (lane < 16) ? 0 : 8;    // A 16x32: lanes>=16 hold K 8..15,24..31
        const int boff = (lane < 16) ? 0 : 8;    // B 32x16: lanes>=16 hold K 8..15,24..31
        v16h afr[4], bfr[2];
#pragma unroll
        for (int mi = 0; mi < 4; ++mi) {
            const _Float16* p = &lA[buf][(waveM * 64 + mi * 16 + rsel) * BK];
            FragCvt cv;
            cv.u[0] = *(const u32x4*)(p + aoff);
            cv.u[1] = *(const u32x4*)(p + aoff + 16);
            afr[mi] = cv.h;
        }
#pragma unroll
        for (int ni = 0; ni < 2; ++ni) {
            const _Float16* p = &lB[buf][(waveN * 32 + ni * 16 + rsel) * BK + boff];
            FragCvt cv;
            cv.u[0] = *(const u32x4*)(p);
            cv.u[1] = *(const u32x4*)(p + 16);
            bfr[ni] = cv.h;
        }
#pragma unroll
        for (int mi = 0; mi < 4; ++mi)
#pragma unroll
            for (int ni = 0; ni < 2; ++ni)
                acc[mi][ni] = __builtin_amdgcn_wmma_f32_16x16x32_f16(
                    false, afr[mi], false, bfr[ni], (short)0, acc[mi][ni],
                    false, false);
    };

    const int NK = K / BK;

#if ASYNC_LDS
    // ---- depth-4 software pipeline over ASYNCcnt (in-order completion) ----
    {
        const int PRE = (NK < 3) ? NK : 3;
        for (int t = 0; t < PRE; ++t) stage1(t & 3, t * BK);
#pragma unroll 1
        for (int kt = 0; kt < NK; ++kt) {
            // own tile kt complete; leave up to 2 tiles (8 ops) in flight
            if (kt + 2 < NK)
                asm volatile("s_wait_asynccnt 0x8" ::: "memory");
            else if (kt + 1 < NK)
                asm volatile("s_wait_asynccnt 0x4" ::: "memory");
            else
                asm volatile("s_wait_asynccnt 0x0" ::: "memory");
            __syncthreads();   // tile kt resident for the whole workgroup
            compute(kt & 3);
            if (kt + 3 < NK) stage1((kt + 3) & 3, (kt + 3) * BK);
        }
    }
#else
    // ---- fallback: global->reg->LDS double buffer ----
    {
        auto stage2 = [&](int buf) {
#pragma unroll
            for (int i = 0; i < 2; ++i) {
                int chunk = tid + i * 256;
                int r = chunk >> 2, c = chunk & 3;
                *(u32x4*)&lA[buf][r * BK + c * 8] = ra[i];
                *(u32x4*)&lB[buf][r * BK + c * 8] = rb[i];
            }
        };
        stage1(0, 0);
        stage2(0);
        __syncthreads();
#pragma unroll 1
        for (int kt = 0; kt < NK; ++kt) {
            const int cur = kt & 1;
            if (kt + 1 < NK) stage1(cur ^ 1, (kt + 1) * BK);
            compute(cur);
            if (kt + 1 < NK) stage2(cur ^ 1);
            __syncthreads();
        }
    }
#endif

    // 16x16 f32 C/D layout: VGPR r -> M = r (+8 for lanes 16..31), N = lane%16
    const int crow = (lane < 16) ? 0 : 8;
    const int ccol = lane & 15;
#pragma unroll
    for (int mi = 0; mi < 4; ++mi)
#pragma unroll
        for (int ni = 0; ni < 2; ++ni) {
            const int gn = n0 + waveN * 32 + ni * 16 + ccol;
            const int gm = m0 + waveM * 64 + mi * 16 + crow;
            if (CT) {
                // transposed store: lane's 8 rows are contiguous along m
#pragma unroll
                for (int r = 0; r < 8; ++r)
                    C[(long long)gn * ldc + (gm + r)] = (OutT)(alpha * acc[mi][ni][r]);
            } else {
#pragma unroll
                for (int r = 0; r < 8; ++r)
                    C[(long long)(gm + r) * ldc + gn] = (OutT)(alpha * acc[mi][ni][r]);
            }
        }
}

// Row softmax, in place on f16 probabilities; one block per row of 2048.
// Each thread owns 8 contiguous halves -> one b128 load + one b128 store.
__global__ __launch_bounds__(256)
void softmax_rows_f16(_Float16* __restrict__ P, float scale)
{
    const int tid = threadIdx.x;
    const int lane = tid & 31, wave = tid >> 5;
    _Float16* row = P + (long long)blockIdx.x * 2048;

    Half8 pk;
    pk.u = ((const u32x4*)row)[tid];

    float v[8];
    float m = -3.40282347e38f;
#pragma unroll
    for (int i = 0; i < 8; ++i) {
        v[i] = (float)pk.h[i] * scale;
        m = fmaxf(m, v[i]);
    }
#pragma unroll
    for (int off = 16; off > 0; off >>= 1)
        m = fmaxf(m, __shfl_xor(m, off, 32));

    __shared__ float smax[8], ssum[8];
    if (lane == 0) smax[wave] = m;
    __syncthreads();
    m = smax[0];
#pragma unroll
    for (int w = 1; w < 8; ++w) m = fmaxf(m, smax[w]);

    float s = 0.f;
#pragma unroll
    for (int i = 0; i < 8; ++i) { v[i] = __expf(v[i] - m); s += v[i]; }
#pragma unroll
    for (int off = 16; off > 0; off >>= 1)
        s += __shfl_xor(s, off, 32);
    if (lane == 0) ssum[wave] = s;
    __syncthreads();
    s = 0.f;
#pragma unroll
    for (int w = 0; w < 8; ++w) s += ssum[w];

    const float inv = 1.f / s;
#pragma unroll
    for (int i = 0; i < 8; ++i)
        pk.h[i] = (_Float16)(v[i] * inv);
    ((u32x4*)row)[tid] = pk.u;
}

// fp32 -> f16 down-convert, vectorized x4.
__global__ __launch_bounds__(256)
void cvt_f32_to_f16(const float* __restrict__ x, _Float16* __restrict__ y, long long n4)
{
    long long i = (long long)blockIdx.x * blockDim.x + threadIdx.x;
    const long long stride = (long long)gridDim.x * blockDim.x;
    for (; i < n4; i += stride) {
        f32x4 f = ((const f32x4*)x)[i];
        v4h h;
        h[0] = (_Float16)f[0]; h[1] = (_Float16)f[1];
        h[2] = (_Float16)f[2]; h[3] = (_Float16)f[3];
        ((v4h*)y)[i] = h;
    }
}

extern "C" void kernel_launch(void* const* d_in, const int* in_sizes, int n_in,
                              void* d_out, int out_size, void* d_ws, size_t ws_size,
                              hipStream_t stream)
{
    (void)in_sizes; (void)n_in; (void)out_size; (void)ws_size;

    const float* q  = (const float*)d_in[0];
    const float* k  = (const float*)d_in[1];
    const float* v  = (const float*)d_in[2];
    const float* Qw = (const float*)d_in[3];
    const float* Kw = (const float*)d_in[4];
    const float* Vw = (const float*)d_in[5];
    float* out = (float*)d_out;

    const int  Bn = 4, S = 2048, E = 1024;
    const int  T  = Bn * S;                     // 8192 tokens
    const long long H = (long long)T * E;       // 8,388,608 elements
    const long long W = (long long)E * E;       // 1,048,576 elements

    // workspace layout (f16 elements)
    _Float16* ws = (_Float16*)d_ws;
    _Float16* xq = ws;          // H : f16 query
    _Float16* xk = xq + H;      // H : f16 key
    _Float16* xv = xk + H;      // H : f16 value
    _Float16* wq = xv + H;      // W : f16 Q_W
    _Float16* wk = wq + W;      // W
    _Float16* wv = wk + W;      // W
    _Float16* Qp = wv + W;      // H : projected Q  [token][e]
    _Float16* Kp = Qp + H;      // H : projected K  [token][e]
    _Float16* Vt = Kp + H;      // H : projected V, TRANSPOSED [e][token]
    _Float16* P  = Vt + H;      // Bn*S*S : scores -> probabilities (in place)

    const dim3 cblk(256);

    // 1) down-convert inputs + weights once (~150MB total, HBM-bound)
    cvt_f32_to_f16<<<2048, cblk, 0, stream>>>(q,  xq, H / 4);
    cvt_f32_to_f16<<<2048, cblk, 0, stream>>>(k,  xk, H / 4);
    cvt_f32_to_f16<<<2048, cblk, 0, stream>>>(v,  xv, H / 4);
    cvt_f32_to_f16<<<1024, cblk, 0, stream>>>(Qw, wq, W / 4);
    cvt_f32_to_f16<<<1024, cblk, 0, stream>>>(Kw, wk, W / 4);
    cvt_f32_to_f16<<<1024, cblk, 0, stream>>>(Vw, wv, W / 4);

    // 2) QKV projections: y = x @ W.T (M=8192, N=1024, K=1024)
    //    V's output is written transposed so P@V can use the K-contiguous loader.
    {
        dim3 grid(T / BM, E / BN, 1);
        gemm_wmma_f16<_Float16, false><<<grid, cblk, 0, stream>>>(
            xq, wq, Qp, E, E, E, E, 0, 0, 0, 1.f);
        gemm_wmma_f16<_Float16, false><<<grid, cblk, 0, stream>>>(
            xk, wk, Kp, E, E, E, E, 0, 0, 0, 1.f);
        gemm_wmma_f16<_Float16, true><<<grid, cblk, 0, stream>>>(
            xv, wv, Vt, E, E, E, /*ldc(T)=*/T, 0, 0, 0, 1.f);
    }

    // 3) scores = Q @ K.T per batch (M=N=2048, K=1024); 1/sqrt(E) applied in softmax
    {
        dim3 grid(S / BM, S / BN, Bn);
        gemm_wmma_f16<_Float16, false><<<grid, cblk, 0, stream>>>(
            Qp, Kp, P, E, E, E, S,
            (long long)S * E, (long long)S * E, (long long)S * S, 1.f);
    }

    // 4) softmax rows (scale = 1/32), f16 in place
    softmax_rows_f16<<<Bn * S, cblk, 0, stream>>>(P, 1.f / 32.f);

    // 5) out = P @ V per batch: A = P [q][s'], B = Vt [e][token] (K-contiguous)
    //    M=2048, N=1024, K=2048; fp32 output
    {
        dim3 grid(S / BM, E / BN, Bn);
        gemm_wmma_f16<float, false><<<grid, cblk, 0, stream>>>(
            P, Vt, out, S, S, /*ldb=*/T, E,
            (long long)S * S, /*sB=*/(long long)S, (long long)S * E, 1.f);
    }
}
